// MaxSimilarityMatcher_77129022702103
// MI455X (gfx1250) — matrix operand order repaired and checked by reading, not compiled, not run on hardware
//
#include <hip/hip_runtime.h>
#include <hip/hip_bf16.h>

// ---------------- problem constants ----------------
constexpr int BB = 4;      // batch
constexpr int NN = 4096;   // keypoints A
constexpr int MM = 4096;   // keypoints B
constexpr int CC = 256;    // channels
constexpr int NPOW = 4;    // steerer order
#define INV_TEMP 20.0f
#define THRESH   0.01f

// ---------------- WMMA types ----------------
typedef __attribute__((ext_vector_type(16))) __bf16 v16bf;
typedef __attribute__((ext_vector_type(8)))  float  v8f;

union FragU {
  v16bf v;
  uint4 q[2];
};

// ---------------- helpers ----------------
__device__ __forceinline__ unsigned short f2bf(float f) {
  unsigned u = __float_as_uint(f);
  unsigned r = u + 0x7FFFu + ((u >> 16) & 1u);   // round-to-nearest-even
  return (unsigned short)(r >> 16);
}

// monotone float<->uint mapping (total order incl. negatives)
__device__ __forceinline__ unsigned fmono(float f) {
  unsigned u = __float_as_uint(f);
  return u ^ (unsigned)(((int)u >> 31) | 0x80000000);
}
__device__ __forceinline__ float fdemono(unsigned m) {
  unsigned u = (m & 0x80000000u) ? (m ^ 0x80000000u) : ~m;
  return __uint_as_float(u);
}

__device__ __forceinline__ float blockReduceSum(float v, float* red) {
  const int t = threadIdx.x;
  red[t] = v; __syncthreads();
  for (int s = 128; s > 0; s >>= 1) {
    if (t < s) red[t] += red[t + s];
    __syncthreads();
  }
  float r = red[0]; __syncthreads();
  return r;
}

// ---------------- K0: zero stats ----------------
__global__ void msm_zero_u32(unsigned* __restrict__ p, int count) {
  int i = blockIdx.x * 256 + threadIdx.x;
  if (i < count) p[i] = 0u;
}

// ---------------- K1: normalize B -> bf16 ----------------
__global__ __launch_bounds__(256)
void msm_prep_B(const float* __restrict__ dB, unsigned short* __restrict__ dBn) {
  __shared__ float red[256];
  const int b = blockIdx.y, m = blockIdx.x, t = threadIdx.x;
  const size_t off = ((size_t)b * MM + m) * CC;
  float v = dB[off + t];
  float ss = blockReduceSum(v * v, red);
  dBn[off + t] = f2bf(v * rsqrtf(ss));
}

// ---------------- K2: steer A (4 powers) -> normalized bf16 ----------------
__global__ __launch_bounds__(256)
void msm_steer(const float* __restrict__ dA, const float* __restrict__ S,
               unsigned short* __restrict__ dAall) {
  __shared__ float cur[CC];
  __shared__ float red[256];
  const int b = blockIdx.y, n = blockIdx.x, t = threadIdx.x;
  cur[t] = dA[((size_t)b * NN + n) * CC + t];
  __syncthreads();
  for (int p = 0; p < NPOW; p++) {
    if (p > 0) {
      float acc = 0.f;
      #pragma unroll 8
      for (int c = 0; c < CC; c++) acc = fmaf(cur[c], S[(size_t)c * CC + t], acc);
      __syncthreads();
      cur[t] = acc;
      __syncthreads();
    }
    float ss = blockReduceSum(cur[t] * cur[t], red);
    dAall[(((size_t)p * BB + b) * NN + n) * CC + t] = f2bf(cur[t] * rsqrtf(ss));
  }
}

// ---------------- K3: WMMA GEMM, max over powers, fused row/col max ----------------
// 128x128 output tile per workgroup; 8 waves, each wave owns 32 rows x 64 cols
// = 2x4 sub-tiles of 16x16, K = 256 in 8 steps of 32 (bf16).
__global__ __launch_bounds__(256)
void msm_corr_gemm(const unsigned short* __restrict__ dAall,   // [4][B][N][C] bf16
                   const unsigned short* __restrict__ dBn,     // [B][M][C] bf16
                   float* __restrict__ corr,                   // [B][N][M] f32
                   unsigned* __restrict__ rowMaxU,
                   unsigned* __restrict__ colMaxU) {
  const int b  = blockIdx.z;
  const int n0 = blockIdx.y * 128;
  const int m0 = blockIdx.x * 128;
  const int lane = threadIdx.x & 31;
  const int wave = threadIdx.x >> 5;
  const int wr = wave >> 1;            // 0..3
  const int wc = wave & 1;             // 0..1
  const int rbase = n0 + wr * 32;
  const int cbase = m0 + wc * 64;
  const int half = lane >> 4;          // 0|1
  const int l15  = lane & 15;

  v8f cmax[2][4];
  #pragma unroll
  for (int i = 0; i < 2; i++)
    #pragma unroll
    for (int j = 0; j < 4; j++)
      #pragma unroll
      for (int e = 0; e < 8; e++) cmax[i][j][e] = -3.4e38f;

  #pragma unroll 1
  for (int p = 0; p < NPOW; p++) {
    const unsigned short* Ab = dAall + ((size_t)(p * BB + b) * NN) * CC;
    v8f acc[2][4];
    #pragma unroll
    for (int i = 0; i < 2; i++)
      #pragma unroll
      for (int j = 0; j < 4; j++)
        #pragma unroll
        for (int e = 0; e < 8; e++) acc[i][j][e] = 0.f;

    #pragma unroll
    for (int kt = 0; kt < 8; kt++) {
      const int k0 = kt * 32;
      // A fragments: ISA 16-bit A 16x32 layout.
      // lanes 0-15: row=l15, K chunks {k0..k0+7, k0+16..k0+23}
      // lanes 16-31: row=l15, K chunks {k0+8..k0+15, k0+24..k0+31}
      FragU af[2];
      #pragma unroll
      for (int mi = 0; mi < 2; mi++) {
        const unsigned short* pr = Ab + (size_t)(rbase + mi * 16 + l15) * CC;
        const int ka = k0 + (half ? 8 : 0);
        const int kb = k0 + (half ? 24 : 16);
        af[mi].q[0] = *(const uint4*)(pr + ka);
        af[mi].q[1] = *(const uint4*)(pr + kb);
      }
      // B fragments: 32x16 (KxN): lane = column l15; lanes 0-15 K chunks {k0..k0+7, k0+16..k0+23},
      // lanes 16-31 K chunks {k0+8..k0+15, k0+24..k0+31}. dBn row-major [M][C] => contiguous in K.
      FragU bfr[4];
      #pragma unroll
      for (int ni = 0; ni < 4; ni++) {
        const unsigned short* pc =
            dBn + ((size_t)b * MM + (cbase + ni * 16 + l15)) * CC + k0 + (half ? 8 : 0);
        bfr[ni].q[0] = *(const uint4*)(pc);
        bfr[ni].q[1] = *(const uint4*)(pc + 16);
      }
      #pragma unroll
      for (int mi = 0; mi < 2; mi++)
        #pragma unroll
        for (int ni = 0; ni < 4; ni++)
          acc[mi][ni] = __builtin_amdgcn_wmma_f32_16x16x32_bf16(
              false, af[mi].v, false, bfr[ni].v, (short)0, acc[mi][ni],
              false, false);
    }
    // running elementwise max over steering powers
    #pragma unroll
    for (int i = 0; i < 2; i++)
      #pragma unroll
      for (int j = 0; j < 4; j++)
        #pragma unroll
        for (int e = 0; e < 8; e++) cmax[i][j][e] = fmaxf(cmax[i][j][e], acc[i][j][e]);
  }

  // Write scaled corr + fused row/col maxima.
  // C/D layout: VGPR e, lanes 0-15 -> (row0+e, col=l15); lanes 16-31 -> (row0+8+e).
  #pragma unroll
  for (int mi = 0; mi < 2; mi++) {
    #pragma unroll
    for (int ni = 0; ni < 4; ni++) {
      const int row0 = rbase + mi * 16 + half * 8;
      const int col  = cbase + ni * 16 + l15;
      float sv[8];
      float cmx = -3.4e38f;
      #pragma unroll
      for (int e = 0; e < 8; e++) {
        sv[e] = cmax[mi][ni][e] * INV_TEMP;
        corr[((size_t)b * NN + row0 + e) * MM + col] = sv[e];
        cmx = fmaxf(cmx, sv[e]);
      }
      // row max: reduce across the 16 lanes of this half
      #pragma unroll
      for (int e = 0; e < 8; e++) {
        float r = sv[e];
        r = fmaxf(r, __shfl_xor(r, 1, 16));
        r = fmaxf(r, __shfl_xor(r, 2, 16));
        r = fmaxf(r, __shfl_xor(r, 4, 16));
        r = fmaxf(r, __shfl_xor(r, 8, 16));
        if (l15 == 0) atomicMax(&rowMaxU[(size_t)b * NN + row0 + e], fmono(r));
      }
      // col max: combine the two halves (same col), one atomic from half 0
      float c2 = fmaxf(cmx, __shfl_xor(cmx, 16, 32));
      if (half == 0) atomicMax(&colMaxU[(size_t)b * MM + col], fmono(c2));
    }
  }
}

// ---------------- K4: row exp-sums (no atomics) ----------------
__global__ __launch_bounds__(256)
void msm_row_expsum(const float* __restrict__ corr, const unsigned* __restrict__ rowMaxU,
                    float* __restrict__ rowSum) {
  __shared__ float red[256];
  const int b = blockIdx.y, n = blockIdx.x, t = threadIdx.x;
  const float rm = fdemono(rowMaxU[(size_t)b * NN + n]);
  const float* r = corr + ((size_t)b * NN + n) * MM;
  float s = 0.f;
  #pragma unroll
  for (int i = 0; i < MM / 256; i++) s += __expf(r[i * 256 + t] - rm);
  float tot = blockReduceSum(s, red);
  if (t == 0) rowSum[(size_t)b * NN + n] = tot;
}

// ---------------- K5a: col exp-sum partials (deterministic) ----------------
__global__ __launch_bounds__(256)
void msm_col_expsum_part(const float* __restrict__ corr, const unsigned* __restrict__ colMaxU,
                         float* __restrict__ colPart) {
  const int b = blockIdx.z, chunk = blockIdx.y, t = threadIdx.x;
  const int col = blockIdx.x * 256 + t;
  const float cm = fdemono(colMaxU[(size_t)b * MM + col]);
  float s = 0.f;
  for (int r = 0; r < 256; r++) {
    int n = chunk * 256 + r;
    s += __expf(corr[((size_t)b * NN + n) * MM + col] - cm);
  }
  colPart[(size_t)chunk * (BB * MM) + (size_t)b * MM + col] = s;
}

// ---------------- K5b: reduce col partials ----------------
__global__ __launch_bounds__(256)
void msm_col_expsum_reduce(const float* __restrict__ colPart, float* __restrict__ colSum) {
  const int i = blockIdx.x * 256 + threadIdx.x;   // over BB*MM
  float s = 0.f;
  #pragma unroll
  for (int c = 0; c < NN / 256; c++) s += colPart[(size_t)c * (BB * MM) + i];
  colSum[i] = s;
}

// ---------------- K6: P = softmax_col * softmax_row (in place) ----------------
__global__ __launch_bounds__(256)
void msm_compute_P(float* __restrict__ corr,
                   const unsigned* __restrict__ rowMaxU, const float* __restrict__ rowSum,
                   const unsigned* __restrict__ colMaxU, const float* __restrict__ colSum) {
  const int b = blockIdx.z, n = blockIdx.y;
  const int col = blockIdx.x * 256 + threadIdx.x;
  const float rm = fdemono(rowMaxU[(size_t)b * NN + n]);
  const float rs = rowSum[(size_t)b * NN + n];
  const float cm = fdemono(colMaxU[(size_t)b * MM + col]);
  const float cs = colSum[(size_t)b * MM + col];
  const size_t idx = ((size_t)b * NN + n) * MM + col;
  const float c = corr[idx];
  corr[idx] = (__expf(c - rm) / rs) * (__expf(c - cm) / cs);
}

// ---------------- K7: P row max + first-occurrence argmax ----------------
__global__ __launch_bounds__(256)
void msm_p_rowmax(const float* __restrict__ Pm, float* __restrict__ PrMax,
                  int* __restrict__ bestCol) {
  __shared__ float sv[256];
  __shared__ int   si[256];
  const int b = blockIdx.y, n = blockIdx.x, t = threadIdx.x;
  const float* r = Pm + ((size_t)b * NN + n) * MM;
  float bv = -1.f; int bi = 0x7fffffff;
  #pragma unroll
  for (int i = 0; i < MM / 256; i++) {
    int m = i * 256 + t;
    float v = r[m];
    if (v > bv || (v == bv && m < bi)) { bv = v; bi = m; }
  }
  sv[t] = bv; si[t] = bi; __syncthreads();
  for (int s = 128; s > 0; s >>= 1) {
    if (t < s) {
      if (sv[t + s] > sv[t] || (sv[t + s] == sv[t] && si[t + s] < si[t])) {
        sv[t] = sv[t + s]; si[t] = si[t + s];
      }
    }
    __syncthreads();
  }
  if (t == 0) {
    PrMax[(size_t)b * NN + n] = sv[0];
    bestCol[(size_t)b * NN + n] = si[0];
  }
}

// ---------------- K8: P col max (P>=0 so raw-bit atomicMax is order-safe) ----------
__global__ __launch_bounds__(256)
void msm_p_colmax(const float* __restrict__ Pm, unsigned* __restrict__ PcMaxU) {
  const int b = blockIdx.z, chunk = blockIdx.y, t = threadIdx.x;
  const int col = blockIdx.x * 256 + t;
  float s = 0.f;
  for (int r = 0; r < 256; r++) {
    int n = chunk * 256 + r;
    s = fmaxf(s, Pm[((size_t)b * NN + n) * MM + col]);
  }
  atomicMax(&PcMaxU[(size_t)b * MM + col], __float_as_uint(s));
}

// ---------------- K9: mutual-NN mask, match extraction ----------------
__global__ __launch_bounds__(256)
void msm_finalize(float* __restrict__ Pm, const float* __restrict__ PrMax,
                  const unsigned* __restrict__ PcMaxU, const int* __restrict__ bestCol,
                  const float* __restrict__ kA, const float* __restrict__ kB,
                  float* __restrict__ mA, float* __restrict__ mB,
                  float* __restrict__ validOut) {
  __shared__ int anyRed[256];
  const int b = blockIdx.y, n = blockIdx.x, t = threadIdx.x;
  const size_t roff = ((size_t)b * NN + n) * MM;
  const float prm = PrMax[(size_t)b * NN + n];
  int any = 0;
  #pragma unroll
  for (int i = 0; i < MM / 256; i++) {
    int m = i * 256 + t;
    float v = Pm[roff + m];
    float pcm = __uint_as_float(PcMaxU[(size_t)b * MM + m]);
    int keep = (v == prm) && (v == pcm) && (v > THRESH);
    Pm[roff + m] = keep ? v : 0.0f;
    any |= keep;
  }
  anyRed[t] = any; __syncthreads();
  for (int s = 128; s > 0; s >>= 1) {
    if (t < s) anyRed[t] |= anyRed[t + s];
    __syncthreads();
  }
  if (t == 0) {
    const size_t ri = (size_t)b * NN + n;
    int bc = bestCol[ri];
    mA[ri * 2 + 0] = kA[ri * 2 + 0];
    mA[ri * 2 + 1] = kA[ri * 2 + 1];
    mB[ri * 2 + 0] = kB[((size_t)b * MM + bc) * 2 + 0];
    mB[ri * 2 + 1] = kB[((size_t)b * MM + bc) * 2 + 1];
    validOut[ri] = anyRed[0] ? 1.0f : 0.0f;
  }
}

// ---------------- host launcher ----------------
extern "C" void kernel_launch(void* const* d_in, const int* in_sizes, int n_in,
                              void* d_out, int out_size, void* d_ws, size_t ws_size,
                              hipStream_t stream) {
  (void)in_sizes; (void)n_in; (void)out_size; (void)ws_size;

  const float* kA = (const float*)d_in[0];   // [B,N,2]
  const float* dA = (const float*)d_in[1];   // [B,N,C]
  const float* kB = (const float*)d_in[2];   // [B,M,2]
  const float* dB = (const float*)d_in[3];   // [B,M,C]
  const float* S  = (const float*)d_in[4];   // [C,C]

  // --- workspace layout ---
  char* ws = (char*)d_ws;
  unsigned short* dBn   = (unsigned short*)ws;                       // B*M*C bf16 = 8 MiB
  unsigned short* dAall = (unsigned short*)(ws + (size_t)BB * MM * CC * 2); // 4*B*N*C bf16 = 32 MiB
  unsigned* statbase = (unsigned*)(ws + (size_t)BB * MM * CC * 2
                                      + (size_t)NPOW * BB * NN * CC * 2);
  unsigned* rowMaxU = statbase;                       // B*N
  unsigned* colMaxU = rowMaxU + (size_t)BB * NN;      // B*M
  float*    rowSum  = (float*)(colMaxU + (size_t)BB * MM);  // B*N
  float*    colSum  = rowSum + (size_t)BB * NN;             // B*M
  unsigned* PcMaxU  = (unsigned*)(colSum + (size_t)BB * MM);// B*M
  float*    PrMax   = (float*)(PcMaxU + (size_t)BB * MM);   // B*N
  int*      bestCol = (int*)(PrMax + (size_t)BB * NN);      // B*N
  float*    colPart = (float*)(bestCol + (size_t)BB * NN);  // 16 * B*M

  // --- output layout (flat concat of the tuple) ---
  float* out   = (float*)d_out;
  float* Pm    = out;                                  // [B,N,M]  (also holds corr first)
  float* mA    = out + (size_t)BB * NN * MM;           // [B,N,2]
  float* mB    = mA + (size_t)BB * NN * 2;             // [B,N,2]
  float* valid = mB + (size_t)BB * NN * 2;             // [B,N]

  // K0: zero the 5 contiguous accumulator arrays (rowMaxU..PcMaxU)
  {
    int cnt = 5 * BB * NN;   // NN == MM so each of the five arrays is BB*4096 u32
    msm_zero_u32<<<(cnt + 255) / 256, 256, 0, stream>>>(statbase, cnt);
  }
  // K1: normalize B descriptors -> bf16
  msm_prep_B<<<dim3(MM, BB), 256, 0, stream>>>(dB, dBn);
  // K2: steered + normalized A descriptors for 4 powers -> bf16
  msm_steer<<<dim3(NN, BB), 256, 0, stream>>>(dA, S, dAall);
  // K3: WMMA correlation GEMM with max over powers; corr -> Pm region
  msm_corr_gemm<<<dim3(MM / 128, NN / 128, BB), 256, 0, stream>>>(
      dAall, dBn, Pm, rowMaxU, colMaxU);
  // K4/K5: softmax denominators
  msm_row_expsum<<<dim3(NN, BB), 256, 0, stream>>>(Pm, rowMaxU, rowSum);
  msm_col_expsum_part<<<dim3(MM / 256, NN / 256, BB), 256, 0, stream>>>(Pm, colMaxU, colPart);
  msm_col_expsum_reduce<<<(BB * MM) / 256, 256, 0, stream>>>(colPart, colSum);
  // K6: dual softmax (in place)
  msm_compute_P<<<dim3(MM / 256, NN, BB), 256, 0, stream>>>(Pm, rowMaxU, rowSum, colMaxU, colSum);
  // K7/K8: P row/col maxima (+argmax)
  msm_p_rowmax<<<dim3(NN, BB), 256, 0, stream>>>(Pm, PrMax, bestCol);
  msm_p_colmax<<<dim3(MM / 256, NN / 256, BB), 256, 0, stream>>>(Pm, PcMaxU);
  // K9: mutual-NN mask, matches, validity
  msm_finalize<<<dim3(NN, BB), 256, 0, stream>>>(Pm, PrMax, PcMaxU, bestCol,
                                                 kA, kB, mA, mB, valid);
}
